// PointerDecoder_5265629905044
// MI455X (gfx1250) — matrix operand order repaired and checked by reading, not compiled, not run on hardware
//
#include <hip/hip_runtime.h>

#define Bn 16
#define Sn 512
#define En 256
#define Hn 512
#define Tn 32

typedef __attribute__((ext_vector_type(16))) __bf16 v16bf;
typedef __attribute__((ext_vector_type(8)))  float  v8f;

static __device__ __forceinline__ float sigm(float v)   { return 1.f / (1.f + __expf(-v)); }
static __device__ __forceinline__ float tanhft(float v) { float e = __expf(2.f * v); return 1.f - 2.f / (e + 1.f); }

// ---------------------------------------------------------------------------
// Kernel 1: ctx[b,o,s] = b_ctx[o] + sum_h W_ctx[o,h] * context[b,s,h]
// One wave32 per (16 x 64) output strip: one o-tile x four s-tiles.
// bf16 WMMA, f32 accumulate, fully unrolled K (16 steps of 32):
//   - A fragment loaded once per k-step, reused by 4 WMMAs (4 accumulators)
//   - native (__bf16) casts -> v_cvt_pk_bf16_f32
// A fragment (16x32, M=o,K=h): lane L -> m=L&15; elems e: K = 16*(L>>4)+e
// B fragment (32x16, K=h,N=s): lane L -> n=L&15; elems e: K = 16*(L>>4)+e
// Both map to contiguous runs of W_ctx / context rows (K = h contiguous).
// ---------------------------------------------------------------------------
__global__ __launch_bounds__(256) void ctx_gemm_kernel(
    const float* __restrict__ context,   // (B,S,H)
    const float* __restrict__ W_ctx,     // (H,H)
    const float* __restrict__ b_ctx,     // (H)
    float* __restrict__ ctx)             // (B,H,S)
{
  const int wave = (blockIdx.x << 3) + (threadIdx.x >> 5);
  const int lane = threadIdx.x & 31;
  const int OT = Hn / 16, SQ = Sn / 64;          // s handled 4 tiles (64 cols) per wave
  const int b  = wave / (OT * SQ);
  const int rm = wave % (OT * SQ);
  const int ot = rm / SQ, sq = rm % SQ;
  const int m  = lane & 15, hi = lane >> 4;

  const float* arow = W_ctx + (size_t)(ot * 16 + m) * Hn;
  // B column n = lane&15 within tile ti -> context row  s = sq*64 + ti*16 + m
  const float* brow0 = context + ((size_t)b * Sn + sq * 64 + m) * Hn;
  const float* brow1 = brow0 + (size_t)16 * Hn;
  const float* brow2 = brow0 + (size_t)32 * Hn;
  const float* brow3 = brow0 + (size_t)48 * Hn;

  v8f acc0 = {}, acc1 = {}, acc2 = {}, acc3 = {};

#pragma unroll
  for (int kt = 0; kt < Hn / 32; ++kt) {
    const int ka = kt * 32 + 16 * hi;    // A fragment K offsets
    const int kb = kt * 32 + 16 * hi;    // B fragment K offsets
    v16bf a, b0, b1, b2, b3;
#pragma unroll
    for (int e = 0; e < 8; ++e) {
      a[e]     = (__bf16)arow[ka + e];
      a[8 + e] = (__bf16)arow[ka + 8 + e];
    }
#pragma unroll
    for (int e = 0; e < 16; ++e) {
      b0[e] = (__bf16)brow0[kb + e];
      b1[e] = (__bf16)brow1[kb + e];
      b2[e] = (__bf16)brow2[kb + e];
      b3[e] = (__bf16)brow3[kb + e];
    }
    acc0 = __builtin_amdgcn_wmma_f32_16x16x32_bf16(false, a, false, b0, (short)0, acc0, false, false);
    acc1 = __builtin_amdgcn_wmma_f32_16x16x32_bf16(false, a, false, b1, (short)0, acc1, false, false);
    acc2 = __builtin_amdgcn_wmma_f32_16x16x32_bf16(false, a, false, b2, (short)0, acc2, false, false);
    acc3 = __builtin_amdgcn_wmma_f32_16x16x32_bf16(false, a, false, b3, (short)0, acc3, false, false);
  }

  // C/D layout: VGPR r -> M = r + 8*(lane>>4), N = lane&15
#pragma unroll
  for (int r = 0; r < 8; ++r) {
    const int o = ot * 16 + r + 8 * hi;
    float* drow = ctx + ((size_t)b * Hn + o) * Sn + sq * 64 + m;
    const float bias = b_ctx[o];
    drow[0]  = acc0[r] + bias;
    drow[16] = acc1[r] + bias;
    drow[32] = acc2[r] + bias;
    drow[48] = acc3[r] + bias;
  }
}

// ---------------------------------------------------------------------------
// Kernel 2: persistent decoder — one workgroup (8 wave32s) per batch element,
// all T=32 steps in-kernel, state in LDS, only workgroup barriers.
// ---------------------------------------------------------------------------
__global__ __launch_bounds__(256) void decoder_kernel(
    const int*   __restrict__ sentence,   // (S,B)
    const float* __restrict__ dec_in,     // (B,E)
    const float* __restrict__ h0,         // (B,H)
    const float* __restrict__ c0,         // (B,H)
    const float* __restrict__ emb,        // (VOCAB,E)
    const float* __restrict__ W_ih, const float* __restrict__ b_ih,   // (4H,E),(4H)
    const float* __restrict__ W_hh, const float* __restrict__ b_hh,   // (4H,H),(4H)
    const float* __restrict__ W_ho, const float* __restrict__ b_ho,   // (H,2H),(H)
    const float* __restrict__ W_in, const float* __restrict__ b_in,   // (H,H),(H)
    const float* __restrict__ Vg,                                     // (H)
    const float* __restrict__ ctx,                                    // (B,H,S)
    float* __restrict__ out_alphas,  // (B,T,S)
    float* __restrict__ out_ptrs,    // (B,T)
    float* __restrict__ out_h,       // (B,H)
    float* __restrict__ out_c)       // (B,H)
{
  __shared__ float x[En], h[Hn], c[Hn], ht[Hn], inp[Hn], hid[Hn];
  __shared__ float mask[Sn], score[Sn], alpha[Sn], Vs[Hn];
  __shared__ float gates[4 * Hn];
  __shared__ float redv[256];
  __shared__ int   redi[256];
  __shared__ float s_gmax, s_gsum;
  __shared__ int   s_gidx;

  const int tid  = threadIdx.x, b = blockIdx.x;
  const int wave = tid >> 5, lane = tid & 31;
  const float* ctxb = ctx + (size_t)b * Hn * Sn;

  for (int i = tid; i < En; i += 256) x[i] = dec_in[b * En + i];
  for (int i = tid; i < Hn; i += 256) { h[i] = h0[b * Hn + i]; c[i] = c0[b * Hn + i]; Vs[i] = Vg[i]; }
  for (int i = tid; i < Sn; i += 256) mask[i] = 1.f;
  __syncthreads();

  for (int t = 0; t < Tn; ++t) {
    // ---- gates = W_ih @ x + W_hh @ h + b_ih + b_hh  (4H rows, 256/wave) ----
    for (int r = 0; r < 256; ++r) {
      const int j = wave * 256 + r;
      const float4* wi = (const float4*)(W_ih + (size_t)j * En);
      const float4* wh = (const float4*)(W_hh + (size_t)j * Hn);
      float acc = 0.f;
#pragma unroll
      for (int q0 = 0; q0 < En / 4; q0 += 32) {
        float4 w = wi[q0 + lane]; const int k = 4 * (q0 + lane);
        acc += w.x * x[k] + w.y * x[k + 1] + w.z * x[k + 2] + w.w * x[k + 3];
      }
#pragma unroll
      for (int q0 = 0; q0 < Hn / 4; q0 += 32) {
        float4 w = wh[q0 + lane]; const int k = 4 * (q0 + lane);
        acc += w.x * h[k] + w.y * h[k + 1] + w.z * h[k + 2] + w.w * h[k + 3];
      }
#pragma unroll
      for (int mm = 16; mm; mm >>= 1) acc += __shfl_xor(acc, mm, 32);
      if (lane == 0) gates[j] = acc + b_ih[j] + b_hh[j];
    }
    __syncthreads();

    // ---- LSTM cell (c updated in place; ht produced) ----
    for (int i = tid; i < Hn; i += 256) {
      const float ig = sigm(gates[i]);
      const float fg = sigm(gates[Hn + i]);
      const float gg = tanhft(gates[2 * Hn + i]);
      const float og = sigm(gates[3 * Hn + i]);
      const float cn = fg * c[i] + ig * gg;
      c[i]  = cn;
      ht[i] = og * tanhft(cn);
    }
    __syncthreads();

    // ---- inp = W_in @ ht + b_in ----
    for (int r = 0; r < 64; ++r) {
      const int j = wave * 64 + r;
      const float4* wr = (const float4*)(W_in + (size_t)j * Hn);
      float acc = 0.f;
#pragma unroll
      for (int q0 = 0; q0 < Hn / 4; q0 += 32) {
        float4 w = wr[q0 + lane]; const int k = 4 * (q0 + lane);
        acc += w.x * ht[k] + w.y * ht[k + 1] + w.z * ht[k + 2] + w.w * ht[k + 3];
      }
#pragma unroll
      for (int mm = 16; mm; mm >>= 1) acc += __shfl_xor(acc, mm, 32);
      if (lane == 0) inp[j] = acc + b_in[j];
    }
    __syncthreads();

    // ---- scores[s] = sum_h V[h]*tanh(inp[h]+ctx[b,h,s]); mask to -inf ----
    for (int s = tid; s < Sn; s += 256) {           // lane<->s : coalesced columns
      const float* cc = ctxb + s;
      float acc = 0.f;
      for (int k = 0; k < Hn; ++k)
        acc += Vs[k] * tanhft(inp[k] + cc[(size_t)k * Sn]);
      score[s] = (mask[s] > 0.f) ? acc : -3.4e38f;
    }
    __syncthreads();

    // ---- max + argmax (first-index tie break) ----
    {
      float lm = -3.4e38f; int li = Sn;
      for (int s = tid; s < Sn; s += 256) {
        const float v = score[s];
        if (v > lm) { lm = v; li = s; }
      }
      redv[tid] = lm; redi[tid] = li;
      __syncthreads();
      for (int off = 128; off > 0; off >>= 1) {
        if (tid < off) {
          const float v2 = redv[tid + off]; const int i2 = redi[tid + off];
          if (v2 > redv[tid] || (v2 == redv[tid] && i2 < redi[tid])) { redv[tid] = v2; redi[tid] = i2; }
        }
        __syncthreads();
      }
      if (tid == 0) { s_gmax = redv[0]; s_gidx = redi[0]; }
      __syncthreads();
    }

    // ---- softmax: exp, sum, normalize, emit alphas ----
    {
      const float gmax = s_gmax;
      float ls = 0.f;
      for (int s = tid; s < Sn; s += 256) {
        const float e = (mask[s] > 0.f) ? __expf(score[s] - gmax) : 0.f;
        alpha[s] = e; ls += e;
      }
      redv[tid] = ls;
      __syncthreads();
      for (int off = 128; off > 0; off >>= 1) {
        if (tid < off) redv[tid] += redv[tid + off];
        __syncthreads();
      }
      if (tid == 0) s_gsum = redv[0];
      __syncthreads();
      const float inv = 1.f / s_gsum;
      for (int s = tid; s < Sn; s += 256) {
        const float a = alpha[s] * inv;
        alpha[s] = a;
        out_alphas[((size_t)b * Tn + t) * Sn + s] = a;
      }
    }
    __syncthreads();

    // ---- hid = ctx[b] @ alpha  (rows contiguous) ----
    for (int r = 0; r < 64; ++r) {
      const int o = wave * 64 + r;
      const float4* cr = (const float4*)(ctxb + (size_t)o * Sn);
      float acc = 0.f;
#pragma unroll
      for (int q0 = 0; q0 < Sn / 4; q0 += 32) {
        float4 w = cr[q0 + lane]; const int k = 4 * (q0 + lane);
        acc += w.x * alpha[k] + w.y * alpha[k + 1] + w.z * alpha[k + 2] + w.w * alpha[k + 3];
      }
#pragma unroll
      for (int mm = 16; mm; mm >>= 1) acc += __shfl_xor(acc, mm, 32);
      if (lane == 0) hid[o] = acc;
    }
    __syncthreads();

    // ---- h = tanh(W_ho @ [hid;ht] + b_ho) ----
    for (int r = 0; r < 64; ++r) {
      const int o = wave * 64 + r;
      const float4* wr = (const float4*)(W_ho + (size_t)o * (2 * Hn));
      float acc = 0.f;
#pragma unroll
      for (int q0 = 0; q0 < Hn / 4; q0 += 32) {
        float4 w = wr[q0 + lane]; const int k = 4 * (q0 + lane);
        acc += w.x * hid[k] + w.y * hid[k + 1] + w.z * hid[k + 2] + w.w * hid[k + 3];
      }
#pragma unroll
      for (int q0 = 0; q0 < Hn / 4; q0 += 32) {
        float4 w = wr[Hn / 4 + q0 + lane]; const int k = 4 * (q0 + lane);
        acc += w.x * ht[k] + w.y * ht[k + 1] + w.z * ht[k + 2] + w.w * ht[k + 3];
      }
#pragma unroll
      for (int mm = 16; mm; mm >>= 1) acc += __shfl_xor(acc, mm, 32);
      if (lane == 0) h[o] = tanhft(acc + b_ho[o]);
    }
    __syncthreads();

    // ---- greedy pointer: mask update + embedding gather ----
    const int gidx = s_gidx;
    if (tid == 0) {
      mask[gidx] = 0.f;
      out_ptrs[b * Tn + t] = (float)gidx;
    }
    const int tok = sentence[gidx * Bn + b];   // sentence is (S,B)
    for (int i = tid; i < En; i += 256) x[i] = emb[(size_t)tok * En + i];
    __syncthreads();
  }

  for (int i = tid; i < Hn; i += 256) { out_h[b * Hn + i] = h[i]; out_c[b * Hn + i] = c[i]; }
}

extern "C" void kernel_launch(void* const* d_in, const int* in_sizes, int n_in,
                              void* d_out, int out_size, void* d_ws, size_t ws_size,
                              hipStream_t stream) {
  (void)in_sizes; (void)n_in; (void)out_size; (void)ws_size;
  const int*   sentence      = (const int*)  d_in[0];
  // d_in[1] = sent_lengths (unused by reference computation)
  const float* decoder_input = (const float*)d_in[2];
  const float* hidden_h      = (const float*)d_in[3];
  const float* hidden_c      = (const float*)d_in[4];
  const float* context       = (const float*)d_in[5];
  const float* emb           = (const float*)d_in[6];
  const float* W_ih  = (const float*)d_in[7];
  const float* b_ih  = (const float*)d_in[8];
  const float* W_hh  = (const float*)d_in[9];
  const float* b_hh  = (const float*)d_in[10];
  const float* W_ho  = (const float*)d_in[11];
  const float* b_ho  = (const float*)d_in[12];
  const float* W_in  = (const float*)d_in[13];
  const float* b_in  = (const float*)d_in[14];
  const float* W_ctx = (const float*)d_in[15];
  const float* b_ctx = (const float*)d_in[16];
  const float* V     = (const float*)d_in[17];

  float* ctx = (float*)d_ws;  // (B,H,S) f32 = 16 MiB scratch

  // 16 * 32 o-tiles * 8 s-quads = 4096 waves, 8 waves/block -> 512 blocks
  const int waves = Bn * (Hn / 16) * (Sn / 64);
  ctx_gemm_kernel<<<dim3(waves / 8), dim3(256), 0, stream>>>(context, W_ctx, b_ctx, ctx);

  float* out        = (float*)d_out;
  float* out_alphas = out;                                  // B*T*S
  float* out_ptrs   = out_alphas + (size_t)Bn * Tn * Sn;    // B*T
  float* out_h      = out_ptrs + Bn * Tn;                   // B*H
  float* out_c      = out_h + Bn * Hn;                      // B*H

  decoder_kernel<<<dim3(Bn), dim3(256), 0, stream>>>(
      sentence, decoder_input, hidden_h, hidden_c, emb,
      W_ih, b_ih, W_hh, b_hh, W_ho, b_ho, W_in, b_in, V, ctx,
      out_alphas, out_ptrs, out_h, out_c);
}
